// SynthesisBlock_65558380806670
// MI455X (gfx1250) — matrix operand order repaired and checked by reading, not compiled, not run on hardware
//
#include <hip/hip_runtime.h>
#include <hip/hip_bf16.h>

// ---------------------------------------------------------------------------
// StyleGAN2 synthesis block for MI455X (gfx1250, wave32, WMMA bf16).
// Heavy convs run as implicit GEMM with V_WMMA_F32_16X16X32_BF16.
// Workgroup tile 128M x 128N, 8 waves (4x2), each wave 32x64 (8 WMMA/step),
// double-buffered LDS with register staging, 1 barrier per K-step.
// ---------------------------------------------------------------------------

typedef __attribute__((ext_vector_type(16))) __bf16 v16bf;
typedef __attribute__((ext_vector_type(8)))  float  v8f;

#define INV_SQRT512 0.04419417382415922f
#define LRELU_GAIN  1.4142135623730951f
#define CLAMP_V     256.0f

// ---------------------------------------------------------------- affines ---
__global__ void affine_kernel(const float* __restrict__ ws,
                              const float* __restrict__ aw0, const float* __restrict__ ab0,
                              const float* __restrict__ aw1, const float* __restrict__ ab1,
                              const float* __restrict__ awr, const float* __restrict__ abr,
                              float* __restrict__ sbuf) {
    __shared__ float red[256];
    int o = blockIdx.x, b = blockIdx.y, sel = blockIdx.z;
    const float* aw = (sel == 0) ? aw0 : ((sel == 1) ? aw1 : awr);
    const float* ab = (sel == 0) ? ab0 : ((sel == 1) ? ab1 : abr);
    const float* wsp = ws + (size_t)(b * 3 + sel) * 512;
    int t = threadIdx.x;
    float acc = wsp[t] * aw[(size_t)o * 512 + t]
              + wsp[t + 256] * aw[(size_t)o * 512 + t + 256];
    red[t] = acc; __syncthreads();
    for (int s = 128; s > 0; s >>= 1) {
        if (t < s) red[t] += red[t + s];
        __syncthreads();
    }
    if (t == 0) {
        float sv = red[0] * INV_SQRT512 + ab[o];
        if (sel == 2) sv *= INV_SQRT512;
        sbuf[sel * 4096 + b * 512 + o] = sv;
    }
}

// ----------------------------------------------- per-(o,i) weight sq-sums ---
__global__ void w2sum_kernel(const float* __restrict__ w0,
                             const float* __restrict__ w1,
                             float* __restrict__ o0, float* __restrict__ o1) {
    int idx = blockIdx.x * 256 + threadIdx.x;          // o*512 + i
    const float* w = (blockIdx.y == 0) ? w0 : w1;
    float* o = (blockIdx.y == 0) ? o0 : o1;
    const float* p = w + (size_t)idx * 9;
    float s = 0.f;
#pragma unroll
    for (int k = 0; k < 9; ++k) s += p[k] * p[k];
    o[idx] = s;
}

// -------------------------------------------------------------- demod -------
__global__ void demod_kernel(const float* __restrict__ sbuf,
                             const float* __restrict__ w2s0,
                             const float* __restrict__ w2s1,
                             float* __restrict__ d0, float* __restrict__ d1) {
    __shared__ float red[256];
    int o = blockIdx.x, b = blockIdx.y, sel = blockIdx.z;
    const float* s  = sbuf + sel * 4096 + b * 512;
    const float* w2 = ((sel == 0) ? w2s0 : w2s1) + (size_t)o * 512;
    int t = threadIdx.x;
    float a = s[t] * s[t] * w2[t] + s[t + 256] * s[t + 256] * w2[t + 256];
    red[t] = a; __syncthreads();
    for (int st = 128; st > 0; st >>= 1) {
        if (t < st) red[t] += red[t + st];
        __syncthreads();
    }
    if (t == 0) ((sel == 0) ? d0 : d1)[b * 512 + o] = rsqrtf(red[0] + 1e-8f);
}

// -------------------------------------- scale input by s0, bf16 NHWC --------
__global__ void scale_input_kernel(const float* __restrict__ x,
                                   const float* __restrict__ sbuf,
                                   __bf16* __restrict__ xs) {
    int idx = blockIdx.x * 256 + threadIdx.x;          // (b,c,h,w) w fastest
    int b  = idx >> 19;
    int c  = (idx >> 10) & 511;
    int hw = idx & 1023;
    float v = x[idx] * sbuf[b * 512 + c];              // s0 at sbuf base
    xs[(((size_t)(b << 10)) + hw) * 512 + c] = (__bf16)v;
}

// ---------------- conv0 phase weights: (convT stride2 3x3) ∘ blur4 ----------
__global__ void prep_w0_kernel(const float* __restrict__ w0,
                               __bf16* __restrict__ w0p) {
    int idx = blockIdx.x * 256 + threadIdx.x;          // o*512 + ci
    int o = idx >> 9, ci = idx & 511;
    const float* w = w0 + (size_t)idx * 9;
    float wl[9];
#pragma unroll
    for (int k = 0; k < 9; ++k) wl[k] = w[k];
    const float f[4] = {0.125f, 0.375f, 0.375f, 0.125f};
    float C6[6][6];
    for (int dy = -2; dy <= 3; ++dy)
        for (int dx = -2; dx <= 3; ++dx) {
            float s = 0.f;
            for (int ty = 0; ty < 4; ++ty) {
                int wy = dy + ty - 1; if (wy < 0 || wy > 2) continue;
                for (int tx = 0; tx < 4; ++tx) {
                    int wx = dx + tx - 1; if (wx < 0 || wx > 2) continue;
                    s += f[ty] * f[tx] * wl[wy * 3 + wx];
                }
            }
            C6[dy + 2][dx + 2] = 4.0f * s;
        }
    for (int py = 0; py < 2; ++py)
        for (int px = 0; px < 2; ++px)
            for (int m = 0; m < 3; ++m)
                for (int n = 0; n < 3; ++n) {
                    int dy = py ? (3 - 2 * m) : (2 - 2 * m);
                    int dx = px ? (3 - 2 * n) : (2 - 2 * n);
                    w0p[((size_t)((py * 2 + px) * 512 + o)) * 4608
                        + (size_t)(m * 3 + n) * 512 + ci] = (__bf16)C6[dy + 2][dx + 2];
                }
}

// ------------------------------------------- repack weight1 to bf16 [o][k] --
__global__ void prep_w1_kernel(const float* __restrict__ w1,
                               __bf16* __restrict__ w1b) {
    int idx = blockIdx.x * 256 + threadIdx.x;          // o*512 + ci
    int o = idx >> 9, ci = idx & 511;
    const float* w = w1 + (size_t)idx * 9;
#pragma unroll
    for (int kk = 0; kk < 9; ++kk)
        w1b[(size_t)o * 4608 + (size_t)kk * 512 + ci] = (__bf16)w[kk];
}

// --------------------------------------------------------------- WMMA conv --
// MODE 0: conv0 phase (input WxW=32x32), epilogue d0/noise/bias/lrelu, *s1,
//         write bf16 NHWC into x0s quadrant (2jy+py, 2jx+px).
// MODE 1: conv1 (input 64x64), epilogue d1/noise/bias/lrelu,
//         write f32 NCHW into d_out x region.
template <int W, int MODE>
__global__ void conv_wmma_kernel(const __bf16* __restrict__ act,
                                 const __bf16* __restrict__ wmat,
                                 const float* __restrict__ dmod,
                                 const float* __restrict__ bias,
                                 const float* __restrict__ noise,
                                 const float* __restrict__ nsp,
                                 const float* __restrict__ s1,
                                 __bf16* __restrict__ out_bf,
                                 float* __restrict__ out_f,
                                 int py, int px) {
    constexpr int LOGW  = (W == 32) ? 5 : 6;
    constexpr int LOGWW = 2 * LOGW;
    __shared__ __align__(16) __bf16 lA[2][128][32];   // [buf][m][k]  8KB each
    __shared__ __align__(16) __bf16 lB[2][128][32];   // [buf][n][k]  8KB each

    int tid  = threadIdx.x;
    int lane = tid & 31;
    int wid  = tid >> 5;
    int wm   = wid & 3;              // M sub-tile (0..3) -> 32 rows each
    int wn   = wid >> 2;             // N sub-tile (0..1) -> 64 cols each
    int lrow = lane & 15;
    int hi   = lane >> 4;            // lane group
    int mbase = blockIdx.y * 128;
    int nbase = blockIdx.x * 128;
    float ns = nsp[0];

    // staging assignments (all 256 threads stage 32B of A and 32B of B)
    int arow   = tid >> 1;           // 0..127
    int achunk = (tid & 1) << 4;     // 0 or 16 (bf16 units)
    int bpx    = tid >> 1;           // 0..127
    int bchunk = (tid & 1) << 4;     // 0 or 16

    // hoisted pixel decode for the B stager (loop-invariant)
    int ngB = nbase + bpx;
    int bB  = ngB >> LOGWW;
    int rmB = ngB & ((1 << LOGWW) - 1);
    int jyB = rmB >> LOGW;
    int jxB = rmB & (W - 1);

    const __bf16* aBase = wmat + (size_t)(mbase + arow) * 4608 + achunk;
    const __bf16* bPix  = act + (((size_t)(bB * W + jyB)) * W + jxB) * 512 + bchunk;

    uint4 ra0, ra1, rb0, rb1;
    auto fetch = [&](int s) {
        const __bf16* ap = aBase + (size_t)s * 32;
        ra0 = *(const uint4*)ap;
        ra1 = *(const uint4*)(ap + 8);
        int pos = s >> 4;
        int ky = pos / 3, kx = pos - 3 * ky;
        int iy = jyB + ky - 1, ix = jxB + kx - 1;
        rb0 = make_uint4(0u, 0u, 0u, 0u);
        rb1 = rb0;
        if ((unsigned)iy < (unsigned)W && (unsigned)ix < (unsigned)W) {
            const __bf16* bp = bPix + (((ky - 1) * W + (kx - 1)) << 9) + ((s & 15) << 5);
            rb0 = *(const uint4*)bp;
            rb1 = *(const uint4*)(bp + 8);
        }
    };
    auto stage = [&](int buf) {
        *(uint4*)&lA[buf][arow][achunk]      = ra0;
        *(uint4*)&lA[buf][arow][achunk + 8]  = ra1;
        *(uint4*)&lB[buf][bpx][bchunk]       = rb0;
        *(uint4*)&lB[buf][bpx][bchunk + 8]   = rb1;
    };

    v8f z = {0.f,0.f,0.f,0.f,0.f,0.f,0.f,0.f};
    v8f c[2][4] = {{z, z, z, z}, {z, z, z, z}};

    fetch(0);
    stage(0);

    for (int s = 0; s < 144; ++s) {
        __syncthreads();
        int cur = s & 1;
        bool more = (s + 1 < 144);
        if (more) fetch(s + 1);                        // global loads in flight
        if (s + 2 < 144)
            __builtin_prefetch(aBase + (size_t)(s + 2) * 32, 0, 0);

        union F { v16bf v; uint4 q[2]; } a[2], b[4];
#pragma unroll
        for (int i = 0; i < 2; ++i) {
            a[i].q[0] = *(const uint4*)&lA[cur][wm * 32 + i * 16 + lrow][hi * 8];
            a[i].q[1] = *(const uint4*)&lA[cur][wm * 32 + i * 16 + lrow][16 + hi * 8];
        }
#pragma unroll
        for (int j = 0; j < 4; ++j) {
            b[j].q[0] = *(const uint4*)&lB[cur][wn * 64 + j * 16 + lrow][hi * 8];
            b[j].q[1] = *(const uint4*)&lB[cur][wn * 64 + j * 16 + lrow][16 + hi * 8];
        }
#pragma unroll
        for (int i = 0; i < 2; ++i)
#pragma unroll
            for (int j = 0; j < 4; ++j)
                c[i][j] = __builtin_amdgcn_wmma_f32_16x16x32_bf16(
                              false, a[i].v, false, b[j].v, (short)0, c[i][j], false, false);

        if (more) stage(cur ^ 1);                      // fill the other buffer
    }

    // ----- epilogue: C layout -> lane l, vgpr v: M = v + 8*hi, N = lrow -----
#pragma unroll
    for (int sm = 0; sm < 2; ++sm) {
#pragma unroll
        for (int sn = 0; sn < 4; ++sn) {
            v8f acc = c[sm][sn];
            int ng = nbase + wn * 64 + sn * 16 + lrow;
            if (MODE == 0) {
                int b = ng >> 10, rem = ng & 1023;
                int jy = rem >> 5, jx = rem & 31;
                int y = 2 * jy + py, xx = 2 * jx + px;
                float nz = noise[(y << 6) + xx] * ns;
                size_t obase = ((((size_t)((b << 6) + y)) << 6) + xx) << 9;
#pragma unroll
                for (int v = 0; v < 8; ++v) {
                    int m = mbase + wm * 32 + sm * 16 + hi * 8 + v;
                    float t = acc[v] * dmod[(b << 9) + m] + nz + bias[m];
                    t = (t >= 0.f ? t : 0.2f * t) * LRELU_GAIN;
                    t = fminf(fmaxf(t, -CLAMP_V), CLAMP_V);
                    out_bf[obase + m] = (__bf16)(t * s1[(b << 9) + m]);
                }
            } else {
                int b = ng >> 12, rem = ng & 4095;     // rem = y*64+x
                float nz = noise[rem] * ns;
#pragma unroll
                for (int v = 0; v < 8; ++v) {
                    int m = mbase + wm * 32 + sm * 16 + hi * 8 + v;
                    float t = acc[v] * dmod[(b << 9) + m] + nz + bias[m];
                    t = (t >= 0.f ? t : 0.2f * t) * LRELU_GAIN;
                    t = fminf(fmaxf(t, -CLAMP_V), CLAMP_V);
                    out_f[(((size_t)((b << 9) + m)) << 12) + rem] = t;
                }
            }
        }
    }
}

// ------------------------------------- toRGB + img 2x upsample + add --------
__global__ void torgb_kernel(const float* __restrict__ xo,
                             const float* __restrict__ img,
                             const float* __restrict__ wrgb,
                             const float* __restrict__ brgb,
                             const float* __restrict__ sbuf,
                             float* __restrict__ imgout) {
    __shared__ float lw[3][512];
    int b = blockIdx.y;
    int tid = threadIdx.x;
    const float* srgb = sbuf + 2 * 4096 + b * 512;
    for (int i = tid; i < 1536; i += 256) {
        int co = i >> 9, ci = i & 511;
        lw[co][ci] = wrgb[co * 512 + ci] * srgb[ci];
    }
    __syncthreads();
    int pix = blockIdx.x * 256 + tid;                  // 0..4095
    const float* xp = xo + ((size_t)b << 21) + pix;    // b*512*4096
    float a0 = 0.f, a1 = 0.f, a2 = 0.f;
    for (int ci = 0; ci < 512; ++ci) {
        float v = xp[(size_t)ci << 12];
        a0 += lw[0][ci] * v;
        a1 += lw[1][ci] * v;
        a2 += lw[2][ci] * v;
    }
    float rc[3];
    rc[0] = fminf(fmaxf(a0 + brgb[0], -CLAMP_V), CLAMP_V);
    rc[1] = fminf(fmaxf(a1 + brgb[1], -CLAMP_V), CLAMP_V);
    rc[2] = fminf(fmaxf(a2 + brgb[2], -CLAMP_V), CLAMP_V);
    int y = pix >> 6, xx = pix & 63;
    int jy[2]; float fy[2];
    if (!(y & 1)) { jy[0] = (y >> 1) - 1; fy[0] = 0.125f; jy[1] = y >> 1;       fy[1] = 0.375f; }
    else          { jy[0] = (y - 1) >> 1; fy[0] = 0.375f; jy[1] = (y + 1) >> 1; fy[1] = 0.125f; }
    int jx[2]; float fx[2];
    if (!(xx & 1)) { jx[0] = (xx >> 1) - 1; fx[0] = 0.125f; jx[1] = xx >> 1;       fx[1] = 0.375f; }
    else           { jx[0] = (xx - 1) >> 1; fx[0] = 0.375f; jx[1] = (xx + 1) >> 1; fx[1] = 0.125f; }
#pragma unroll
    for (int co = 0; co < 3; ++co) {
        const float* ip = img + (size_t)(b * 3 + co) * 1024;
        float u = 0.f;
#pragma unroll
        for (int ty = 0; ty < 2; ++ty) {
            if (jy[ty] < 0 || jy[ty] > 31) continue;
#pragma unroll
            for (int tx = 0; tx < 2; ++tx) {
                if (jx[tx] < 0 || jx[tx] > 31) continue;
                u += fy[ty] * fx[tx] * ip[jy[ty] * 32 + jx[tx]];
            }
        }
        u *= 4.f;
        imgout[(((size_t)(b * 3 + co)) << 12) + pix] = u + rc[co];
    }
}

// ---------------------------------------------------------------------------
extern "C" void kernel_launch(void* const* d_in, const int* in_sizes, int n_in,
                              void* d_out, int out_size, void* d_ws, size_t ws_size,
                              hipStream_t stream) {
    const float* x    = (const float*)d_in[0];
    const float* img  = (const float*)d_in[1];
    const float* ws   = (const float*)d_in[2];
    const float* aw0  = (const float*)d_in[3];
    const float* ab0  = (const float*)d_in[4];
    const float* w0   = (const float*)d_in[5];
    const float* b0   = (const float*)d_in[6];
    const float* ns0  = (const float*)d_in[7];
    const float* aw1  = (const float*)d_in[8];
    const float* ab1  = (const float*)d_in[9];
    const float* w1   = (const float*)d_in[10];
    const float* b1   = (const float*)d_in[11];
    const float* ns1  = (const float*)d_in[12];
    const float* awr  = (const float*)d_in[13];
    const float* abr  = (const float*)d_in[14];
    const float* wrgb = (const float*)d_in[15];
    const float* brgb = (const float*)d_in[16];
    const float* nc0  = (const float*)d_in[17];
    const float* nc1  = (const float*)d_in[18];

    char* wsb = (char*)d_ws;
    float*  sbuf = (float*)(wsb + 0);          // s0|s1|srgb : 3*4096 f32
    float*  d0   = (float*)(wsb + 49152);      // 8*512 f32
    float*  d1   = (float*)(wsb + 65536);
    float*  w2s0 = (float*)(wsb + 81920);      // 512*512 f32
    float*  w2s1 = (float*)(wsb + 1130496);
    __bf16* xs   = (__bf16*)(wsb + 2179072);   // 8*32*32*512 bf16 (NHWC, s0-scaled)
    __bf16* w0p  = (__bf16*)(wsb + 10567680);  // 4 phases * 512*4608 bf16
    __bf16* w1b  = (__bf16*)(wsb + 29442048);  // 512*4608 bf16
    __bf16* x0s  = (__bf16*)(wsb + 34160640);  // 8*64*64*512 bf16 (NHWC, s1-scaled)

    float* xout   = (float*)d_out;             // (8,512,64,64) f32
    float* imgout = (float*)d_out + 16777216;  // (8,3,64,64)  f32

    affine_kernel<<<dim3(512, 8, 3), 256, 0, stream>>>(ws, aw0, ab0, aw1, ab1, awr, abr, sbuf);
    w2sum_kernel<<<dim3(1024, 2), 256, 0, stream>>>(w0, w1, w2s0, w2s1);
    demod_kernel<<<dim3(512, 8, 2), 256, 0, stream>>>(sbuf, w2s0, w2s1, d0, d1);
    scale_input_kernel<<<dim3(16384), 256, 0, stream>>>(x, sbuf, xs);
    prep_w0_kernel<<<dim3(1024), 256, 0, stream>>>(w0, w0p);
    prep_w1_kernel<<<dim3(1024), 256, 0, stream>>>(w1, w1b);

    // conv0 as 4 phase convs: grid N = 8*32*32/128 = 64, M = 512/128 = 4
    for (int p = 0; p < 4; ++p) {
        conv_wmma_kernel<32, 0><<<dim3(64, 4), 256, 0, stream>>>(
            xs, w0p + (size_t)p * 512 * 4608, d0, b0, nc0, ns0,
            sbuf + 4096 /*s1*/, x0s, nullptr, p >> 1, p & 1);
    }
    // conv1: grid N = 8*64*64/128 = 256, M = 4
    conv_wmma_kernel<64, 1><<<dim3(256, 4), 256, 0, stream>>>(
        x0s, w1b, d1, b1, nc1, ns1, nullptr, nullptr, xout, 0, 0);

    torgb_kernel<<<dim3(16, 8), 256, 0, stream>>>(xout, img, wrgb, brgb, sbuf, imgout);
}
